// ChunkedAttention_61117384622564
// MI455X (gfx1250) — compile-verified, matrix operand order checked by reading, not run on hardware
//
#include <hip/hip_runtime.h>

typedef __bf16 bf16;
typedef __attribute__((ext_vector_type(16))) __bf16 v16bf;
typedef __attribute__((ext_vector_type(8)))  __bf16 bf16x8;
typedef __attribute__((ext_vector_type(8)))  float v8f;
typedef __attribute__((ext_vector_type(4)))  float f32x4;
typedef __attribute__((ext_vector_type(4)))  unsigned short us4;
typedef __attribute__((ext_vector_type(8)))  unsigned short us8;

// Problem dims (fixed by reference): B=1, T=2048, D=1024, H=16, hd=64
#define SEQ_T 2048
#define DMODEL 1024
#define NQKV 3072
#define HEADS 16
#define HD 64
#define LSTR 40   // LDS row stride in bf16 elems (32 + 8 pad, keeps 16B alignment)
#define PSTR 40
#define KSTEPS (DMODEL / 32)

__device__ __forceinline__ unsigned short f2bf(float f) {
  unsigned int u = __float_as_uint(f);
  u += 0x7FFFu + ((u >> 16) & 1u);      // round-to-nearest-even
  return (unsigned short)(u >> 16);
}

__device__ __forceinline__ v8f wmma_bf16(v16bf a, v16bf b, v8f c) {
  // (neg_a, A, neg_b, B, c_mod, C, reuse_a, reuse_b)
  return __builtin_amdgcn_wmma_f32_16x16x32_bf16(false, a, false, b, (short)0, c,
                                                 false, false);
}

__device__ __forceinline__ v8f zero8() {
  v8f z;
#pragma unroll
  for (int i = 0; i < 8; ++i) z[i] = 0.f;
  return z;
}

// 16-bit A/B fragment (16x32 / 32x16): lane lm=l%16 selects row(A)/col(B),
// half h=l/16; VGPR0..3 hold K = h*8..h*8+7, VGPR4..7 hold K = 16+h*8..23+h*8.
__device__ __forceinline__ v16bf load_frag(const bf16* base, int h) {
  bf16x8 lo = *(const bf16x8*)(base + h * 8);
  bf16x8 hi = *(const bf16x8*)(base + 16 + h * 8);
  v16bf f;
#pragma unroll
  for (int i = 0; i < 8; ++i) { f[i] = lo[i]; f[i + 8] = hi[i]; }
  return f;
}

// Async DMA of one 128x32 bf16 A-tile (row-major, stride DMODEL) into LDS.
// 512 x 16B chunks, 256 threads -> 2 global_load_async_to_lds_b128 per thread.
// GVS addressing: 64-bit SGPR base + per-lane 32-bit byte offset.
__device__ __forceinline__ void async_tile_a(const unsigned short* __restrict__ gA,
                                             const unsigned short* lbuf,
                                             int tid, int m0, int k0) {
#pragma unroll
  for (int p = 0; p < 2; ++p) {
    int idx = p * 256 + tid;
    int row = idx >> 2;
    int kc = (idx & 3) * 8;
    unsigned loff = (unsigned)(size_t)(lbuf + row * LSTR + kc);       // LDS byte addr
    unsigned goff = (unsigned)(((m0 + row) * DMODEL + k0 + kc) * 2);  // global byte off
    asm volatile("global_load_async_to_lds_b128 %0, %1, %2"
                 :: "v"(loff), "v"(goff), "s"(gA)
                 : "memory");
  }
}

__device__ __forceinline__ void wait_async_le2() {
  asm volatile("s_wait_asynccnt 2" ::: "memory");
}
__device__ __forceinline__ void wait_async_0() {
  asm volatile("s_wait_asynccnt 0" ::: "memory");
}

// ---------------------------------------------------------------- convert
__global__ __launch_bounds__(256) void cvt_f32_bf16(const float* __restrict__ in,
                                                    unsigned short* __restrict__ out,
                                                    int n4) {
  int i = blockIdx.x * blockDim.x + threadIdx.x;
  if (i < n4) {
    f32x4 v = *(const f32x4*)(in + (size_t)i * 4);
    us4 o;
#pragma unroll
    for (int j = 0; j < 4; ++j) o[j] = f2bf(v[j]);
    *(us4*)(out + (size_t)i * 4) = o;
  }
}

// ------------------------------------------------- GEMM1: qkv = x @ W_qkv
// A: bf16 [2048][1024] row-major.  B: bf16 [1024][3072] row-major.
// Double-buffered: A via async DMA to LDS, B via regs (transposed [n][k]).
// Epilogue scatters to qws/kws [H][T][64] and vws [H][64][T] (transposed).
__global__ __launch_bounds__(256) void gemm_qkv(const unsigned short* __restrict__ A,
                                                const unsigned short* __restrict__ B,
                                                unsigned short* __restrict__ qws,
                                                unsigned short* __restrict__ kws,
                                                unsigned short* __restrict__ vws) {
  __shared__ __attribute__((aligned(16))) unsigned short ldsA[2][128 * LSTR];
  __shared__ __attribute__((aligned(16))) unsigned short ldsB[2][128 * LSTR];
  const int tid = threadIdx.x;
  const int n0 = blockIdx.x * 128;
  const int m0 = blockIdx.y * 128;
  const int lane = tid & 31, wave = tid >> 5;
  const int wr = (wave & 3) * 32;               // wave's 32-row strip
  const int wc = (wave >> 2) * 64;              // wave's 64-col strip
  const int lm = lane & 15, h = lane >> 4;
  // per-thread B staging coordinates (2 passes)
  const int bk0 = tid >> 4;                     // 0..15
  const int bk1 = bk0 + 16;                     // 16..31
  const int bn8 = (tid & 15) * 8;

  v8f acc[2][4];
#pragma unroll
  for (int i = 0; i < 2; ++i)
#pragma unroll
    for (int j = 0; j < 4; ++j) acc[i][j] = zero8();

  // prologue: tile 0 in flight
  async_tile_a(A, ldsA[0], tid, m0, 0);
  us8 br0 = *(const us8*)(B + (size_t)bk0 * NQKV + n0 + bn8);
  us8 br1 = *(const us8*)(B + (size_t)bk1 * NQKV + n0 + bn8);

  for (int it = 0; it < KSTEPS; ++it) {
    const int c = it & 1;
    // commit this tile's B regs into LDS (transposed)
#pragma unroll
    for (int i = 0; i < 8; ++i) {
      ldsB[c][(bn8 + i) * LSTR + bk0] = br0[i];
      ldsB[c][(bn8 + i) * LSTR + bk1] = br1[i];
    }
    const bool more = (it + 1 < KSTEPS);
    if (more) {                                  // tile it+1 in flight
      const int kn = (it + 1) * 32;
      async_tile_a(A, ldsA[c ^ 1], tid, m0, kn);
      br0 = *(const us8*)(B + (size_t)(kn + bk0) * NQKV + n0 + bn8);
      br1 = *(const us8*)(B + (size_t)(kn + bk1) * NQKV + n0 + bn8);
    }
    if (more) wait_async_le2(); else wait_async_0();   // own tile-it DMA done
    __syncthreads();                                   // all waves' tile visible

    v16bf af[2], bfr[4];
#pragma unroll
    for (int i = 0; i < 2; ++i)
      af[i] = load_frag((const bf16*)(&ldsA[c][(wr + i * 16 + lm) * LSTR]), h);
#pragma unroll
    for (int j = 0; j < 4; ++j)
      bfr[j] = load_frag((const bf16*)(&ldsB[c][(wc + j * 16 + lm) * LSTR]), h);
#pragma unroll
    for (int i = 0; i < 2; ++i)
#pragma unroll
      for (int j = 0; j < 4; ++j) acc[i][j] = wmma_bf16(af[i], bfr[j], acc[i][j]);
    __syncthreads();                                   // done reading tile it
  }
  // C layout: lane holds col n=lm, rows M = r + 8*h
#pragma unroll
  for (int i = 0; i < 2; ++i)
#pragma unroll
    for (int j = 0; j < 4; ++j)
#pragma unroll
      for (int r = 0; r < 8; ++r) {
        int m = m0 + wr + i * 16 + r + 8 * h;
        int n = n0 + wc + j * 16 + lm;
        unsigned short bv = f2bf(acc[i][j][r]);
        int part = n >> 10;
        int head = (n >> 6) & 15;
        int d = n & 63;
        if (part == 0)
          qws[((size_t)(head << 11) + m) * HD + d] = bv;
        else if (part == 1)
          kws[((size_t)(head << 11) + m) * HD + d] = bv;
        else
          vws[((size_t)(head << 6) + d) * SEQ_T + m] = bv;
      }
}

// ---------------------------------------------- flash attention (causal)
// 1 wave per (head, 16-query tile). 32 keys per iteration.
__global__ __launch_bounds__(128) void attn(const unsigned short* __restrict__ qws,
                                            const unsigned short* __restrict__ kws,
                                            const unsigned short* __restrict__ vws,
                                            unsigned short* __restrict__ yws) {
  __shared__ __attribute__((aligned(16))) unsigned short ldsP[4 * 16 * PSTR];
  const int tid = threadIdx.x;
  const int lane = tid & 31;
  const int wib = tid >> 5;
  const int wid = blockIdx.x * 4 + wib;
  const int head = wid >> 7;                       // 128 q-tiles per head
  const int q0 = (wid & 127) * 16;
  const int lm = lane & 15, h = lane >> 4;

  const unsigned short* Q = qws + (size_t)(head << 11) * HD;
  const unsigned short* K = kws + (size_t)(head << 11) * HD;
  const unsigned short* V = vws + (size_t)(head << 6) * SEQ_T;   // [64][T]
  unsigned short* ldsp = ldsP + wib * 16 * PSTR;

  v16bf qf[2];
  qf[0] = load_frag((const bf16*)(Q + (size_t)(q0 + lm) * HD), h);
  qf[1] = load_frag((const bf16*)(Q + (size_t)(q0 + lm) * HD + 32), h);

  float mi[8], li[8];
  v8f o[4];
#pragma unroll
  for (int r = 0; r < 8; ++r) { mi[r] = -1e30f; li[r] = 0.f; }
#pragma unroll
  for (int g = 0; g < 4; ++g) o[g] = zero8();

  const int nkb = (q0 + 16 + 31) >> 5;
  const float scale = 0.125f;                      // 1/sqrt(64)

  for (int kb = 0; kb < nkb; ++kb) {
    const int kbase = kb * 32;
    if (kb + 1 < nkb) {                            // prefetch next key block
      __builtin_prefetch(K + (size_t)(kbase + 32 + lm) * HD, 0, 1);
      __builtin_prefetch(V + (size_t)lm * SEQ_T + kbase + 32, 0, 1);
    }
    // ---- S = Q K^T : two 16x16 tiles (keys kbase+lm and kbase+16+lm)
    v8f sa = zero8(), sb = zero8();
    {
      v16bf k0f = load_frag((const bf16*)(K + (size_t)(kbase + lm) * HD), h);
      v16bf k1f = load_frag((const bf16*)(K + (size_t)(kbase + lm) * HD + 32), h);
      sa = wmma_bf16(qf[0], k0f, sa);
      sa = wmma_bf16(qf[1], k1f, sa);
      v16bf k2f = load_frag((const bf16*)(K + (size_t)(kbase + 16 + lm) * HD), h);
      v16bf k3f = load_frag((const bf16*)(K + (size_t)(kbase + 16 + lm) * HD + 32), h);
      sb = wmma_bf16(qf[0], k2f, sb);
      sb = wmma_bf16(qf[1], k3f, sb);
    }
    // ---- causal mask + online softmax (row M = r + 8*h, col = key)
    float pa[8], pb[8], rmax[8];
#pragma unroll
    for (int r = 0; r < 8; ++r) {
      int qidx = q0 + r + 8 * h;
      float va = sa[r] * scale;
      float vb = sb[r] * scale;
      if (kbase + lm > qidx) va = -1e30f;
      if (kbase + 16 + lm > qidx) vb = -1e30f;
      pa[r] = va; pb[r] = vb;
      rmax[r] = fmaxf(va, vb);
    }
#pragma unroll
    for (int off = 1; off < 16; off <<= 1)
#pragma unroll
      for (int r = 0; r < 8; ++r)
        rmax[r] = fmaxf(rmax[r], __shfl_xor(rmax[r], off, 32));
    float corr[8], rsum[8];
#pragma unroll
    for (int r = 0; r < 8; ++r) {
      float mnew = fmaxf(mi[r], rmax[r]);
      corr[r] = __expf(mi[r] - mnew);
      mi[r] = mnew;
      pa[r] = __expf(pa[r] - mnew);
      pb[r] = __expf(pb[r] - mnew);
      rsum[r] = pa[r] + pb[r];
    }
#pragma unroll
    for (int off = 1; off < 16; off <<= 1)
#pragma unroll
      for (int r = 0; r < 8; ++r) rsum[r] += __shfl_xor(rsum[r], off, 32);
#pragma unroll
    for (int r = 0; r < 8; ++r) li[r] = li[r] * corr[r] + rsum[r];
    // ---- P: C-layout -> A-layout via per-wave LDS tile
#pragma unroll
    for (int r = 0; r < 8; ++r) {
      int row = r + 8 * h;
      ldsp[row * PSTR + lm] = f2bf(pa[r]);
      ldsp[row * PSTR + 16 + lm] = f2bf(pb[r]);
    }
    // rescale O by correction, then O += P @ V
#pragma unroll
    for (int g = 0; g < 4; ++g)
#pragma unroll
      for (int r = 0; r < 8; ++r) o[g][r] *= corr[r];
    v16bf pf = load_frag((const bf16*)(ldsp + lm * PSTR), h);
#pragma unroll
    for (int g = 0; g < 4; ++g) {
      v16bf vf = load_frag((const bf16*)(V + (size_t)(g * 16 + lm) * SEQ_T + kbase), h);
      o[g] = wmma_bf16(pf, vf, o[g]);
    }
  }
  // normalize and write y (bf16, [T][D], head-major columns)
#pragma unroll
  for (int r = 0; r < 8; ++r) {
    float inv = 1.f / li[r];
    int row = q0 + r + 8 * h;
#pragma unroll
    for (int g = 0; g < 4; ++g)
      yws[(size_t)row * DMODEL + (head << 6) + g * 16 + lm] = f2bf(o[g][r] * inv);
  }
}

// ------------------------------------------------- GEMM2: out = y @ W_o
__global__ __launch_bounds__(256) void gemm_out(const unsigned short* __restrict__ A,
                                                const unsigned short* __restrict__ B,
                                                float* __restrict__ out) {
  __shared__ __attribute__((aligned(16))) unsigned short ldsA[2][128 * LSTR];
  __shared__ __attribute__((aligned(16))) unsigned short ldsB[2][128 * LSTR];
  const int tid = threadIdx.x;
  const int n0 = blockIdx.x * 128;
  const int m0 = blockIdx.y * 128;
  const int lane = tid & 31, wave = tid >> 5;
  const int wr = (wave & 3) * 32;
  const int wc = (wave >> 2) * 64;
  const int lm = lane & 15, h = lane >> 4;
  const int bk0 = tid >> 4;
  const int bk1 = bk0 + 16;
  const int bn8 = (tid & 15) * 8;

  v8f acc[2][4];
#pragma unroll
  for (int i = 0; i < 2; ++i)
#pragma unroll
    for (int j = 0; j < 4; ++j) acc[i][j] = zero8();

  async_tile_a(A, ldsA[0], tid, m0, 0);
  us8 br0 = *(const us8*)(B + (size_t)bk0 * DMODEL + n0 + bn8);
  us8 br1 = *(const us8*)(B + (size_t)bk1 * DMODEL + n0 + bn8);

  for (int it = 0; it < KSTEPS; ++it) {
    const int c = it & 1;
#pragma unroll
    for (int i = 0; i < 8; ++i) {
      ldsB[c][(bn8 + i) * LSTR + bk0] = br0[i];
      ldsB[c][(bn8 + i) * LSTR + bk1] = br1[i];
    }
    const bool more = (it + 1 < KSTEPS);
    if (more) {
      const int kn = (it + 1) * 32;
      async_tile_a(A, ldsA[c ^ 1], tid, m0, kn);
      br0 = *(const us8*)(B + (size_t)(kn + bk0) * DMODEL + n0 + bn8);
      br1 = *(const us8*)(B + (size_t)(kn + bk1) * DMODEL + n0 + bn8);
    }
    if (more) wait_async_le2(); else wait_async_0();
    __syncthreads();

    v16bf af[2], bfr[4];
#pragma unroll
    for (int i = 0; i < 2; ++i)
      af[i] = load_frag((const bf16*)(&ldsA[c][(wr + i * 16 + lm) * LSTR]), h);
#pragma unroll
    for (int j = 0; j < 4; ++j)
      bfr[j] = load_frag((const bf16*)(&ldsB[c][(wc + j * 16 + lm) * LSTR]), h);
#pragma unroll
    for (int i = 0; i < 2; ++i)
#pragma unroll
      for (int j = 0; j < 4; ++j) acc[i][j] = wmma_bf16(af[i], bfr[j], acc[i][j]);
    __syncthreads();
  }
#pragma unroll
  for (int i = 0; i < 2; ++i)
#pragma unroll
    for (int j = 0; j < 4; ++j)
#pragma unroll
      for (int r = 0; r < 8; ++r) {
        int m = m0 + wr + i * 16 + r + 8 * h;
        int n = n0 + wc + j * 16 + lm;
        out[(size_t)m * DMODEL + n] = acc[i][j][r];
      }
}

// ---------------------------------------------------------------- launch
extern "C" void kernel_launch(void* const* d_in, const int* in_sizes, int n_in,
                              void* d_out, int out_size, void* d_ws, size_t ws_size,
                              hipStream_t stream) {
  const float* x = (const float*)d_in[0];      // [1,2048,1024]
  const float* wqkv = (const float*)d_in[1];   // [1024,3072]
  const float* wo = (const float*)d_in[2];     // [1024,1024]
  float* out = (float*)d_out;                  // [1,2048,1024]

  char* ws = (char*)d_ws;
  unsigned short* xb  = (unsigned short*)(ws);                          // 4 MB
  unsigned short* wqb = (unsigned short*)(ws + ((size_t)4 << 20));      // 6 MB
  unsigned short* wob = (unsigned short*)(ws + ((size_t)10 << 20));     // 2 MB
  unsigned short* qws = (unsigned short*)(ws + ((size_t)12 << 20));     // 4 MB
  unsigned short* kws = (unsigned short*)(ws + ((size_t)16 << 20));     // 4 MB
  unsigned short* vws = (unsigned short*)(ws + ((size_t)20 << 20));     // 4 MB
  unsigned short* yws = (unsigned short*)(ws + ((size_t)24 << 20));     // 4 MB

  cvt_f32_bf16<<<(2097152 / 4) / 256, 256, 0, stream>>>(x, xb, 2097152 / 4);
  cvt_f32_bf16<<<(3145728 / 4) / 256, 256, 0, stream>>>(wqkv, wqb, 3145728 / 4);
  cvt_f32_bf16<<<(1048576 / 4) / 256, 256, 0, stream>>>(wo, wob, 1048576 / 4);
  gemm_qkv<<<dim3(NQKV / 128, SEQ_T / 128), 256, 0, stream>>>(xb, wqb, qws, kws, vws);
  attn<<<(HEADS * SEQ_T / 16) / 4, 128, 0, stream>>>(qws, kws, vws, yws);
  gemm_out<<<dim3(DMODEL / 128, SEQ_T / 128), 256, 0, stream>>>(yws, wob, out);
}
